// Molecule3DBias_69655779606854
// MI455X (gfx1250) — matrix operand order repaired and checked by reading, not compiled, not run on hardware
//
#include <hip/hip_runtime.h>

// ---------------- problem constants ----------------
#define Bq 2
#define Nq 512
#define Kq 128
#define Hq 32
#define Eq 768
#define INV_SQRT_2PI 0.3989422804014327f
#define NEG_HALF_LOG2E -0.7213475204444817f   // -0.5 * log2(e)

typedef __attribute__((ext_vector_type(16))) __bf16 v16bf;
typedef __attribute__((ext_vector_type(8)))  __bf16 v8bf;
typedef __attribute__((ext_vector_type(8)))  float  v8f;

// padded LDS row strides (elements) — 16B-aligned, bank-conflict-friendly
#define PSI_STRIDE 136   // 128 + 8
#define W_STRIDE   136
#define ATT_STRIDE 33    // 32 + 1 (floats)

// LDS layout byte offsets
#define OFF_PSI   0
#define OFF_W1    (OFF_PSI  + Nq*PSI_STRIDE*2)        // 139264
#define OFF_W2    (OFF_W1   + Kq*W_STRIDE*2)          // +34816
#define OFF_HID   (OFF_W2   + Hq*W_STRIDE*2)          // +8704
#define OFF_ATT   (OFF_HID  + 8*16*W_STRIDE*2)        // +34816
#define OFF_MEANS (OFF_ATT  + Nq*ATT_STRIDE*4)        // +67584
#define OFF_INVS  (OFF_MEANS + Kq*4)
#define OFF_COEF  (OFF_INVS  + Kq*4)
#define OFF_B1    (OFF_COEF  + Kq*4)
#define OFF_B2    (OFF_B1    + Kq*4)
#define SMEM_TOTAL (OFF_B2   + Hq*4)                  // = 287360 bytes

// Assemble A fragment (16-bit A 16x32 layout, ISA 7.12.2):
// lanes 0-15: k = 0..7 (v0-3) and 16..23 (v4-7); lanes 16-31: +8
__device__ __forceinline__ v16bf load_a_frag(const __bf16* __restrict__ row,
                                             int kb, int half) {
  const v8bf lo = *(const v8bf*)(row + kb * 32 + half * 8);
  const v8bf hi = *(const v8bf*)(row + kb * 32 + 16 + half * 8);
  return __builtin_shufflevector(lo, hi, 0, 1, 2, 3, 4, 5, 6, 7,
                                 8, 9, 10, 11, 12, 13, 14, 15);
}

// Assemble B fragment (16-bit B 32x16 layout): lane = column n,
// lanes 0-15 hold k = kb*32 + 0..7,16..23; lanes 16-31 hold k = kb*32 + 8..15,24..31.
// Column n of B == row h of weight matrix (contiguous over k).
__device__ __forceinline__ v16bf load_b_frag(const __bf16* __restrict__ wrow,
                                             int kb, int half) {
  const v8bf lo = *(const v8bf*)(wrow + kb * 32 + half * 8);
  const v8bf hi = *(const v8bf*)(wrow + kb * 32 + 16 + half * 8);
  return __builtin_shufflevector(lo, hi, 0, 1, 2, 3, 4, 5, 6, 7,
                                 8, 9, 10, 11, 12, 13, 14, 15);
}

// One workgroup per (b,i): psi -> (GEMM1 relu GEMM2) fused in LDS.
__global__ __launch_bounds__(256)
void m3dbias_fused_kernel(const float* __restrict__ pos,
                          const int*   __restrict__ nte,
                          const float* __restrict__ means,
                          const float* __restrict__ stds,
                          const float* __restrict__ mulw,
                          const float* __restrict__ biasw,
                          const float* __restrict__ w1,
                          const float* __restrict__ b1,
                          const float* __restrict__ w2,
                          const float* __restrict__ b2,
                          float* __restrict__ attn_out,
                          float* __restrict__ normed_out,
                          float* __restrict__ pooled_ws) {
  extern __shared__ char smem[];
  __bf16* psi_l   = (__bf16*)(smem + OFF_PSI);
  __bf16* w1_l    = (__bf16*)(smem + OFF_W1);
  __bf16* w2_l    = (__bf16*)(smem + OFF_W2);
  __bf16* hid_l   = (__bf16*)(smem + OFF_HID);
  float*  attn_l  = (float*)(smem + OFF_ATT);
  float*  meanscn_l = (float*)(smem + OFF_MEANS);   // -mean/sigma
  float*  invs_l    = (float*)(smem + OFF_INVS);    // 1/sigma
  float*  coef_l    = (float*)(smem + OFF_COEF);    // 1/(sigma*sqrt(2pi))
  float*  b1_l      = (float*)(smem + OFF_B1);
  float*  b2_l      = (float*)(smem + OFF_B2);

  const int tid  = threadIdx.x;
  const int wv   = tid >> 5;
  const int lane = tid & 31;
  const int half = lane >> 4;
  const int m    = lane & 15;

  const int bi = blockIdx.x;           // 0..B*N-1
  const int b  = bi >> 9;
  const int i  = bi & (Nq - 1);

  // ---------- phase 0: stage weights / params into LDS ----------
  if (tid < Kq) {
    const float s = fabsf(stds[tid]) + 0.01f;
    const float inv = 1.0f / s;
    invs_l[tid]    = inv;
    meanscn_l[tid] = -means[tid] * inv;
    coef_l[tid]    = inv * INV_SQRT_2PI;
    b1_l[tid]      = b1[tid];
  }
  if (tid < Hq) b2_l[tid] = b2[tid];
  for (int idx = tid; idx < Kq * Kq; idx += 256) {
    const int h = idx >> 7, k = idx & (Kq - 1);
    w1_l[h * W_STRIDE + k] = (__bf16)w1[idx];
  }
  for (int idx = tid; idx < Hq * Kq; idx += 256) {
    const int h = idx >> 7, k = idx & (Kq - 1);
    w2_l[h * W_STRIDE + k] = (__bf16)w2[idx];
  }

  const float pix = pos[((size_t)b * Nq + i) * 3 + 0];
  const float piy = pos[((size_t)b * Nq + i) * 3 + 1];
  const float piz = pos[((size_t)b * Nq + i) * 3 + 2];
  __syncthreads();

  // ---------- phase 1: psi into LDS + normed_delta to HBM ----------
  for (int j = tid; j < Nq; j += 256) {
    const float dx = pos[((size_t)b * Nq + j) * 3 + 0] - pix;
    const float dy = pos[((size_t)b * Nq + j) * 3 + 1] - piy;
    const float dz = pos[((size_t)b * Nq + j) * 3 + 2] - piz;
    const float dist = sqrtf(dx * dx + dy * dy + dz * dz);
    const float invd = 1.0f / (dist + 1e-5f);
    const size_t nd = (((size_t)b * Nq + i) * Nq + j) * 3;
    normed_out[nd + 0] = dx * invd;
    normed_out[nd + 1] = dy * invd;
    normed_out[nd + 2] = dz * invd;

    const int e = nte[((size_t)b * Nq + i) * Nq + j];
    const float tr = mulw[e] * dist + biasw[e];

    __bf16* prow = psi_l + j * PSI_STRIDE;
    for (int k8 = 0; k8 < Kq; k8 += 8) {
      v8bf chunk;
#pragma unroll
      for (int t = 0; t < 8; ++t) {
        const int k = k8 + t;
        const float z = fmaf(tr, invs_l[k], meanscn_l[k]);     // (tr-mean)/sigma
        const float p =
            __builtin_amdgcn_exp2f(z * z * NEG_HALF_LOG2E) * coef_l[k];
        chunk[t] = (__bf16)p;
      }
      *(v8bf*)(prow + k8) = chunk;   // one 16B DS store per 8 gaussians
    }
  }
  __syncthreads();

  // ---------- phase 2a: pooled[k] = sum_j psi (waves 0-3) ----------
  if (tid < Kq) {
    float s = 0.0f;
    for (int j = 0; j < Nq; ++j) s += (float)psi_l[j * PSI_STRIDE + tid];
    pooled_ws[((size_t)b * Nq + i) * Kq + tid] = s;
  }

  // ---------- phase 2b: fused WMMA GEMM1 (relu) + GEMM2 ----------
  // loop-invariant operands: w2 B-fragments (8) + biases
  v16bf bw2[8];
#pragma unroll
  for (int ht = 0; ht < 2; ++ht)
#pragma unroll
    for (int kb = 0; kb < 4; ++kb)
      bw2[ht * 4 + kb] = load_b_frag(w2_l + (ht * 16 + m) * W_STRIDE, kb, half);
  float bb1[8];
#pragma unroll
  for (int ht = 0; ht < 8; ++ht) bb1[ht] = b1_l[ht * 16 + m];
  const float bb2_0 = b2_l[m], bb2_1 = b2_l[16 + m];

  __bf16* hw = hid_l + wv * 16 * W_STRIDE;   // per-wave 16x128 scratch
  for (int jt = 0; jt < 4; ++jt) {
    const int j0 = (wv * 4 + jt) * 16;
    v16bf a[4];
#pragma unroll
    for (int kb = 0; kb < 4; ++kb)
      a[kb] = load_a_frag(psi_l + (j0 + m) * PSI_STRIDE, kb, half);

    // GEMM1: hidden = relu(psi @ w1^T + b1)   (8 h-tiles of 16)
#pragma unroll
    for (int ht = 0; ht < 8; ++ht) {
      const int h0 = ht * 16;
      v8f c;
#pragma unroll
      for (int t = 0; t < 8; ++t) c[t] = bb1[ht];
#pragma unroll
      for (int kb = 0; kb < 4; ++kb) {
        const v16bf bf = load_b_frag(w1_l + (h0 + m) * W_STRIDE, kb, half);
        c = __builtin_amdgcn_wmma_f32_16x16x32_bf16(false, a[kb], false, bf,
                                                    (short)0, c, false, false);
      }
#pragma unroll
      for (int r = 0; r < 8; ++r)
        hw[(r + half * 8) * W_STRIDE + h0 + m] = (__bf16)fmaxf(c[r], 0.0f);
    }
    asm volatile("s_wait_dscnt 0" ::: "memory");

    // GEMM2: attn = hidden @ w2^T + b2   (2 h-tiles of 16)
    v16bf a2[4];
#pragma unroll
    for (int kb = 0; kb < 4; ++kb)
      a2[kb] = load_a_frag(hw + m * W_STRIDE, kb, half);
#pragma unroll
    for (int ht = 0; ht < 2; ++ht) {
      const int h0 = ht * 16;
      v8f c;
      const float bb = ht ? bb2_1 : bb2_0;
#pragma unroll
      for (int t = 0; t < 8; ++t) c[t] = bb;
#pragma unroll
      for (int kb = 0; kb < 4; ++kb)
        c = __builtin_amdgcn_wmma_f32_16x16x32_bf16(false, a2[kb], false,
                                                    bw2[ht * 4 + kb],
                                                    (short)0, c, false, false);
#pragma unroll
      for (int r = 0; r < 8; ++r)
        attn_l[(j0 + r + half * 8) * ATT_STRIDE + h0 + m] = c[r];
    }
  }
  __syncthreads();

  // ---------- phase 3: coalesced transposed write (float4 per lane) ----------
  // threads split: tid[7] selects h parity, low 7 bits select a 4-j chunk.
  {
    const int j0 = (tid & 127) * 4;
    const int hp = tid >> 7;                  // 0 or 1
    for (int hh = 0; hh < Hq / 2; ++hh) {
      const int h = hh * 2 + hp;
      float4 v;
      v.x = attn_l[(j0 + 0) * ATT_STRIDE + h];
      v.y = attn_l[(j0 + 1) * ATT_STRIDE + h];
      v.z = attn_l[(j0 + 2) * ATT_STRIDE + h];
      v.w = attn_l[(j0 + 3) * ATT_STRIDE + h];
      const size_t base = (((size_t)b * Hq + h) * Nq + i) * Nq;
      *(float4*)(attn_out + base + j0) = v;
    }
  }
}

// merged = pooled @ proj_w^T + proj_b : (1024 x 768 x 128), one 16x16 tile/wave
__global__ __launch_bounds__(256)
void m3dbias_merged_kernel(const float* __restrict__ pooled,
                           const float* __restrict__ projw,
                           const float* __restrict__ projb,
                           float* __restrict__ out) {
  const int tid  = threadIdx.x;
  const int wv   = tid >> 5;
  const int lane = tid & 31;
  const int half = lane >> 4;
  const int m    = lane & 15;

  const int tile = blockIdx.x * 8 + wv;          // 0..3071
  const int mt = tile & 63;                      // 64 row tiles
  const int et = tile >> 6;                      // 48 col tiles

  const int row = mt * 16 + m;
  const int e   = et * 16 + m;

  const float* __restrict__ prow = pooled + (size_t)row * Kq;
  const float* __restrict__ wrow = projw + (size_t)e * Kq;

  v16bf a[4], bf[4];
#pragma unroll
  for (int kb = 0; kb < 4; ++kb) {
    const int ka = kb * 32 + half * 8;
#pragma unroll
    for (int t = 0; t < 8; ++t) {
      a[kb][t]     = (__bf16)prow[ka + t];
      a[kb][t + 8] = (__bf16)prow[ka + 16 + t];
    }
    const int kbb = kb * 32 + half * 8;
#pragma unroll
    for (int t = 0; t < 8; ++t) {
      bf[kb][t]     = (__bf16)wrow[kbb + t];
      bf[kb][t + 8] = (__bf16)wrow[kbb + 16 + t];
    }
  }

  const float bb = projb[e];
  v8f c;
#pragma unroll
  for (int t = 0; t < 8; ++t) c[t] = bb;
#pragma unroll
  for (int kb = 0; kb < 4; ++kb)
    c = __builtin_amdgcn_wmma_f32_16x16x32_bf16(false, a[kb], false, bf[kb],
                                                (short)0, c, false, false);
#pragma unroll
  for (int r = 0; r < 8; ++r)
    out[(size_t)(mt * 16 + r + half * 8) * Eq + e] = c[r];
}

extern "C" void kernel_launch(void* const* d_in, const int* in_sizes, int n_in,
                              void* d_out, int out_size, void* d_ws, size_t ws_size,
                              hipStream_t stream) {
  (void)in_sizes; (void)n_in; (void)out_size; (void)ws_size;
  const float* pos    = (const float*)d_in[0];
  // d_in[1] = x (unused by reference)
  const int*   nte    = (const int*)d_in[2];
  const float* means  = (const float*)d_in[3];
  const float* stds   = (const float*)d_in[4];
  const float* mulw   = (const float*)d_in[5];
  const float* biasw  = (const float*)d_in[6];
  const float* w1     = (const float*)d_in[7];
  const float* b1     = (const float*)d_in[8];
  const float* w2     = (const float*)d_in[9];
  const float* b2     = (const float*)d_in[10];
  const float* projw  = (const float*)d_in[11];
  const float* projb  = (const float*)d_in[12];

  float* attn_out   = (float*)d_out;                                  // B*H*N*N
  float* merged_out = attn_out + (size_t)Bq * Hq * Nq * Nq;           // B*N*E
  float* normed_out = merged_out + (size_t)Bq * Nq * Eq;              // B*N*N*3

  float* pooled_ws = (float*)d_ws;                                    // B*N*K f32

  hipLaunchKernelGGL(m3dbias_fused_kernel, dim3(Bq * Nq), dim3(256),
                     SMEM_TOTAL, stream,
                     pos, nte, means, stds, mulw, biasw, w1, b1, w2, b2,
                     attn_out, normed_out, pooled_ws);

  hipLaunchKernelGGL(m3dbias_merged_kernel, dim3((64 * 48) / 8), dim3(256),
                     0, stream, pooled_ws, projw, projb, merged_out);
}
